// FrameTransformer_83348135346769
// MI455X (gfx1250) — matrix operand order repaired and checked by reading, not compiled, not run on hardware
//
#include <hip/hip_runtime.h>
#include <math.h>

#define HEADS 4
#define TT 512            // time dimension is 512 everywhere in this model
#define CDIV(a,b) (((a)+(b)-1)/(b))

typedef __attribute__((ext_vector_type(16))) __bf16 v16bf;
typedef __attribute__((ext_vector_type(4)))  __bf16 v4bf;
typedef __attribute__((ext_vector_type(8)))  float  v8f;

enum { FLAG_RESID = 2 };

__device__ __forceinline__ __bf16 tobf(float f) { return (__bf16)f; }

// K index held by element e of a 16-bit A operand (16x32, wave32), matching B:
// lanes 0-15: K0..15, lanes 16-31: K16..31 (contiguous).
__device__ __forceinline__ int a_kidx(int e, int half) {
  return half * 16 + e;
}

// ---- CDNA5 async global->LDS copy (ASYNCcnt path), with portable fallback ----
#if defined(__HIP_DEVICE_COMPILE__) && __has_builtin(__builtin_amdgcn_global_load_async_to_lds_b128)
#define USE_ASYNC_LDS 1
#else
#define USE_ASYNC_LDS 0
#endif

#if USE_ASYNC_LDS
typedef int v4i_vs __attribute__((__vector_size__(16)));
typedef v4i_vs __attribute__((address_space(1)))* gv4i_p;   // global int4*
typedef v4i_vs __attribute__((address_space(3)))* lv4i_p;   // LDS int4*
__device__ __forceinline__ void async_copy16(const __bf16* g, __bf16* l) {
  // flat == global numerically; flat LDS aperture keeps byte offset in low 32 bits (ISA 10.2)
  __builtin_amdgcn_global_load_async_to_lds_b128(
      (gv4i_p)(uintptr_t)g,
      (lv4i_p)(uint32_t)(uintptr_t)l,
      0, 0);
}
__device__ __forceinline__ void async_wait0() {
#if __has_builtin(__builtin_amdgcn_s_wait_asynccnt)
  __builtin_amdgcn_s_wait_asynccnt(0);
#else
  asm volatile("s_wait_asynccnt 0" ::: "memory");
#endif
}
#endif

// ---------------------------------------------------------------------------
// elementwise / normalization kernels (f32)
// ---------------------------------------------------------------------------
__global__ void copy_f32(const float* __restrict__ S, float* __restrict__ D, size_t n) {
  size_t i = (size_t)blockIdx.x * 256 + threadIdx.x;
  if (i < n) D[i] = S[i];
}

// vectorized f32 -> bf16 conversion (flat; used for weights)
__global__ void cvt_bf16_kernel(const float* __restrict__ X, __bf16* __restrict__ Y, size_t n4) {
  size_t i = (size_t)blockIdx.x * 256 + threadIdx.x;
  if (i >= n4) return;
  float4 v = ((const float4*)X)[i];
  v4bf o;
  o[0] = tobf(v.x); o[1] = tobf(v.y); o[2] = tobf(v.z); o[3] = tobf(v.w);
  ((v4bf*)Y)[i] = o;
}

// tiled transpose-convert: X (C, I, T) f32 -> Y (C, T, I) bf16, optional fused sqrelu.
// Makes the GEMM B-operand K-contiguous per N row (2x ds_load_b128 fragment fills).
__global__ void cvt_bf16_t_kernel(const float* __restrict__ X, __bf16* __restrict__ Y,
                                  int I, int sqrelu) {
  const int c  = blockIdx.z;
  const int t0 = blockIdx.x * 32;
  const int i0 = blockIdx.y * 32;
  __shared__ float tile[32][33];
  const float* Xc = X + (size_t)c * I * TT;
  __bf16*      Yc = Y + (size_t)c * TT * I;
  const int tid = threadIdx.x;
  const int col  = tid & 31;      // t offset (coalesced reads)
  const int row4 = tid >> 5;      // 0..7
#pragma unroll
  for (int rr = 0; rr < 4; ++rr) {
    int r = row4 * 4 + rr;        // i offset 0..31
    float v = Xc[(size_t)(i0 + r) * TT + (t0 + col)];
    if (sqrelu) { v = fmaxf(v, 0.f); v *= v; }
    tile[r][col] = v;
  }
  __syncthreads();
  const int tr = tid >> 3;        // t offset 0..31
  const int ic = (tid & 7) * 4;   // i offset, 4 consecutive
  v4bf o;
#pragma unroll
  for (int j = 0; j < 4; ++j) o[j] = tobf(tile[ic + j][tr]);
  *(v4bf*)(Yc + (size_t)(t0 + tr) * I + (i0 + ic)) = o;   // coalesced 8B stores
}

// LayerNorm over the H axis for each (c, t) column; affine per (c, h).
__global__ void frame_norm_kernel(const float* __restrict__ X, const float* __restrict__ Wn,
                                  const float* __restrict__ Bn, float* __restrict__ Y,
                                  int C, int H) {
  int i = blockIdx.x * 256 + threadIdx.x;
  if (i >= C * TT) return;
  int t = i % TT, c = i / TT;
  const float* xc = X + (size_t)c * H * TT + t;
  float s = 0.f, ss = 0.f;
  for (int h = 0; h < H; ++h) { float v = xc[(size_t)h * TT]; s += v; ss += v * v; }
  float mu  = s / (float)H;
  float var = ss / (float)H - mu * mu;
  float inv = rsqrtf(var + 1e-5f);
  float* yc = Y + (size_t)c * H * TT + t;
  for (int h = 0; h < H; ++h)
    yc[(size_t)h * TT] = (xc[(size_t)h * TT] - mu) * inv * Wn[c * H + h] + Bn[c * H + h];
}

// In-place rotary embedding on a (C, F, T) buffer; pairs along d within each head.
__global__ void rotary_kernel(float* __restrict__ X, int C, int F, int hd) {
  size_t total = (size_t)C * (F / 2) * TT;
  size_t i = (size_t)blockIdx.x * 256 + threadIdx.x;
  if (i >= total) return;
  int t = (int)(i % TT);
  size_t j = i / TT;
  int rh = (int)(j % (size_t)(F / 2));
  int c  = (int)(j / (size_t)(F / 2));
  int r  = 2 * rh;          // even row; d parity == row parity since hd is even
  int dd = r % hd;          // even d within head
  float inv = powf(10000.f, -(float)dd / (float)hd);
  float ang = (float)t * inv;
  float sn, cs;
  __sincosf(ang, &sn, &cs);
  size_t base = ((size_t)c * F + r) * TT + t;
  float x1 = X[base], x2 = X[base + TT];
  X[base]      = x1 * cs - x2 * sn;
  X[base + TT] = x2 * cs + x1 * sn;
}

__global__ void softmax_rows(float* __restrict__ S) {
  __shared__ float red[256];
  float* p = S + (size_t)blockIdx.x * TT;
  int tid = threadIdx.x;
  float m = -1e30f;
  for (int i = tid; i < TT; i += 256) m = fmaxf(m, p[i]);
  red[tid] = m; __syncthreads();
  for (int w = 128; w > 0; w >>= 1) { if (tid < w) red[tid] = fmaxf(red[tid], red[tid + w]); __syncthreads(); }
  m = red[0]; __syncthreads();
  float sum = 0.f;
  for (int i = tid; i < TT; i += 256) { float e = __expf(p[i] - m); p[i] = e; sum += e; }
  red[tid] = sum; __syncthreads();
  for (int w = 128; w > 0; w >>= 1) { if (tid < w) red[tid] += red[tid + w]; __syncthreads(); }
  float inv = 1.f / red[0];
  for (int i = tid; i < TT; i += 256) p[i] *= inv;
}

// ---------------------------------------------------------------------------
// per-channel GEMM via v_wmma_f32_16x16x32_bf16 (the dominant FLOP path)
// Out[c](OxT) = W[c](OxI) @ X[c] with X stored TRANSPOSED: Xt (TxI) bf16.
// Block: 256 threads = 8 wave32 -> 2x4 grid of 16x16 WMMA tiles (32x64 block tile).
// Staging: 16-byte async copies to LDS (ASYNCcnt) or uint4 fallback.
// Both fragments are K-contiguous in LDS -> 2x ds_load_b128 each.
// ---------------------------------------------------------------------------
#define ASTR 40   // 80 B row stride (16B multiple)
#define BSTR 40   // 80 B row stride (16B multiple)

__launch_bounds__(256)
__global__ void gemm_mcl_wmma(const __bf16* __restrict__ W, const __bf16* __restrict__ Xt,
                              const float* __restrict__ R, float* __restrict__ Out,
                              int O, int I, int flags) {
  const int c = blockIdx.z;
  const int mBase = blockIdx.y * 32;
  const int nBase = blockIdx.x * 64;
  const __bf16* Wc = W  + (size_t)c * O * I;
  const __bf16* Xc = Xt + (size_t)c * TT * I;     // (T, I) per channel
  const size_t obase = (size_t)c * O * TT;

  __shared__ __align__(16) __bf16 As[32 * ASTR];  // 32(M) x 32(K)
  __shared__ __align__(16) __bf16 Bs[64 * BSTR];  // 64(N) x 32(K)  (K-contiguous per N row)

  const int tid  = threadIdx.x;
  const int lane = tid & 31;
  const int wave = tid >> 5;
  const int wm   = wave >> 2;     // 0..1 (M tile)
  const int wn   = wave & 3;      // 0..3 (N tile)
  const int idx  = lane & 15;
  const int half = lane >> 4;

  // staging: one 16B (8 x bf16) chunk per thread; 4 chunks per 32-K row
  const int ar = tid >> 2, ac = (tid & 3) * 8;    // A tile: threads 0..127 (32 rows)
  const int br = tid >> 2, bc = (tid & 3) * 8;    // B tile: all 256 threads (64 rows)

  v8f acc = {};
  for (int k0 = 0; k0 < I; k0 += 32) {
    __syncthreads();
#if USE_ASYNC_LDS
    if (tid < 128)
      async_copy16(Wc + (size_t)(mBase + ar) * I + (k0 + ac), &As[ar * ASTR + ac]);
    async_copy16(Xc + (size_t)(nBase + br) * I + (k0 + bc), &Bs[br * BSTR + bc]);
    async_wait0();
#else
    if (tid < 128)
      *(uint4*)&As[ar * ASTR + ac] =
          *(const uint4*)(const void*)(Wc + (size_t)(mBase + ar) * I + (k0 + ac));
    *(uint4*)&Bs[br * BSTR + bc] =
        *(const uint4*)(const void*)(Xc + (size_t)(nBase + br) * I + (k0 + bc));
#endif
    __syncthreads();
    v16bf av, bv;
#pragma unroll
    for (int e = 0; e < 16; ++e) {
      av[e] = As[(wm * 16 + idx) * ASTR + a_kidx(e, half)];     // A: lane = M row
      bv[e] = Bs[(wn * 16 + idx) * BSTR + half * 16 + e];       // B: lane = N row, K contiguous
    }
    acc = __builtin_amdgcn_wmma_f32_16x16x32_bf16(false, av, false, bv, (short)0, acc, false, false);
  }
  const int ncol = nBase + wn * 16 + idx;
#pragma unroll
  for (int r = 0; r < 8; ++r) {                   // D: VGPR r -> M = half*8 + r
    int mrow = mBase + wm * 16 + half * 8 + r;
    size_t off = obase + (size_t)mrow * TT + ncol;
    float vo = acc[r];
    if (flags & FLAG_RESID) vo += R[off];
    Out[off] = vo;
  }
}

// ---------------------------------------------------------------------------
// attention: S = (Q^T K) * scale per (channel offset applied by caller, head)
// Q,K stored as (head*hd + d, t); one wave per 16x16 score tile.
// ---------------------------------------------------------------------------
__global__ void attn_qk_wmma(const float* __restrict__ Q, const float* __restrict__ K,
                             float* __restrict__ S, int hd, float scale) {
  const int head  = blockIdx.z;
  const int qBase = blockIdx.y * 16;
  const int kBase = blockIdx.x * 16;
  const int lane  = threadIdx.x;
  const int idx = lane & 15, half = lane >> 4;
  const float* Qh = Q + (size_t)head * hd * TT;
  const float* Kh = K + (size_t)head * hd * TT;
  v8f acc = {};
  for (int d0 = 0; d0 < hd; d0 += 32) {
    v16bf av, bv;
#pragma unroll
    for (int e = 0; e < 16; ++e) {
      int da = d0 + a_kidx(e, half);
      av[e] = (da < hd) ? tobf(Qh[(size_t)da * TT + qBase + idx]) : tobf(0.f);
      int db = d0 + half * 16 + e;
      bv[e] = (db < hd) ? tobf(Kh[(size_t)db * TT + kBase + idx]) : tobf(0.f);
    }
    acc = __builtin_amdgcn_wmma_f32_16x16x32_bf16(false, av, false, bv, (short)0, acc, false, false);
  }
#pragma unroll
  for (int r = 0; r < 8; ++r)
    S[(size_t)head * TT * TT + (size_t)(qBase + half * 8 + r) * TT + kBase + idx] = acc[r] * scale;
}

// A_out[head*hd + d, q] = sum_k P[q,k] * V[head*hd + d, k]
__global__ void attn_pv_wmma(const float* __restrict__ S, const float* __restrict__ V,
                             float* __restrict__ A, int hd) {
  const int head  = blockIdx.z;
  const int qBase = blockIdx.x * 16;
  const int dBase = blockIdx.y * 16;
  const int lane  = threadIdx.x;
  const int idx = lane & 15, half = lane >> 4;
  const float* Sh = S + (size_t)head * TT * TT;
  const float* Vh = V + (size_t)head * hd * TT;
  const int d = dBase + idx;
  v8f acc = {};
  for (int k0 = 0; k0 < TT; k0 += 32) {
    v16bf av, bv;
#pragma unroll
    for (int e = 0; e < 16; ++e) {
      av[e] = tobf(Sh[(size_t)(qBase + idx) * TT + k0 + a_kidx(e, half)]);
      int kb = k0 + half * 16 + e;
      bv[e] = (d < hd) ? tobf(Vh[(size_t)d * TT + kb]) : tobf(0.f);
    }
    acc = __builtin_amdgcn_wmma_f32_16x16x32_bf16(false, av, false, bv, (short)0, acc, false, false);
  }
  if (d < hd) {
#pragma unroll
    for (int r = 0; r < 8; ++r)
      A[((size_t)head * hd + d) * TT + qBase + half * 8 + r] = acc[r];
  }
}

// ---------------------------------------------------------------------------
// conv / residual / resize kernels (tiny FLOP share, plain f32)
// ---------------------------------------------------------------------------
__global__ void conv3x3_kernel(const float* __restrict__ X, const float* __restrict__ W,
                               float* __restrict__ Y, int Cin, int Cout, int H, int Ho,
                               int sH, int sqin) {
  int i = blockIdx.x * 256 + threadIdx.x;
  int total = Cout * Ho * TT;
  if (i >= total) return;
  int t = i % TT, y = (i / TT) % Ho, co = i / (TT * Ho);
  float acc = 0.f;
  for (int ci = 0; ci < Cin; ++ci) {
    const float* Xc = X + (size_t)ci * H * TT;
    const float* Wc = W + ((size_t)co * Cin + ci) * 9;
#pragma unroll
    for (int dy = 0; dy < 3; ++dy) {
      int yy = y * sH + dy - 1;
      if (yy < 0 || yy >= H) continue;
#pragma unroll
      for (int dx = 0; dx < 3; ++dx) {
        int xx = t + dx - 1;
        if (xx < 0 || xx >= TT) continue;
        float v = Xc[(size_t)yy * TT + xx];
        if (sqin) { v = fmaxf(v, 0.f); v *= v; }
        acc += v * Wc[dy * 3 + dx];
      }
    }
  }
  Y[i] = acc;
}

// out = conv1x1(x, idw, stride sH) + h2
__global__ void res_combine_kernel(const float* __restrict__ X, const float* __restrict__ IDW,
                                   const float* __restrict__ H2, float* __restrict__ Y,
                                   int Cin, int Cout, int Hin, int Ho, int sH) {
  int i = blockIdx.x * 256 + threadIdx.x;
  int total = Cout * Ho * TT;
  if (i >= total) return;
  int t = i % TT, y = (i / TT) % Ho, co = i / (TT * Ho);
  float acc = H2[i];
  for (int ci = 0; ci < Cin; ++ci)
    acc += IDW[co * Cin + ci] * X[((size_t)ci * Hin + y * sH) * TT + t];
  Y[i] = acc;
}

__global__ void upsample_h_kernel(const float* __restrict__ X, float* __restrict__ Y,
                                  int C, int H) {
  int i = blockIdx.x * 256 + threadIdx.x;
  int H2 = 2 * H;
  int total = C * H2 * TT;
  if (i >= total) return;
  int t = i % TT, y = (i / TT) % H2, c = i / (TT * H2);
  float pos = (float)y * (float)(H - 1) / (float)(H2 - 1);
  int lo = (int)floorf(pos);
  int hi = lo + 1; if (hi > H - 1) hi = H - 1;
  float fr = pos - (float)lo;
  Y[i] = X[((size_t)c * H + lo) * TT + t] * (1.f - fr) + X[((size_t)c * H + hi) * TT + t] * fr;
}

// final 1x1 projection + edge pad of one extra row
__global__ void out_proj_pad_kernel(const float* __restrict__ Hf, const float* __restrict__ P,
                                    float* __restrict__ O, int Cin, int H) {
  int i = blockIdx.x * 256 + threadIdx.x;
  int Hp = H + 1;
  int total = 2 * Hp * TT;
  if (i >= total) return;
  int t = i % TT, y = (i / TT) % Hp, o = i / (TT * Hp);
  int ys = (y < H) ? y : (H - 1);
  float acc = 0.f;
  for (int c = 0; c < Cin; ++c)
    acc += P[o * Cin + c] * Hf[((size_t)c * H + ys) * TT + t];
  O[i] = acc;
}

// ---------------------------------------------------------------------------
// host-side orchestration
// ---------------------------------------------------------------------------
struct FteP { const float *n1w,*n1b,*qw,*kw,*vw,*ow,*n2w,*n2b,*c1w,*c2w; };
struct ResP { const float *nw,*nb,*c1,*c2,*idw; };
struct Bump { char* base; size_t off; };

static void* balloc_raw(Bump& b, size_t bytes) {
  void* p = b.base + b.off;
  b.off += (bytes + 255) & ~(size_t)255;
  return p;
}
static float* balloc(Bump& b, size_t elems) { return (float*)balloc_raw(b, elems * 4); }

// flat convert (weights)
static __bf16* cvt_to_bf16(const float* X, size_t n, Bump& ws, hipStream_t s) {
  __bf16* Y = (__bf16*)balloc_raw(ws, n * 2);
  cvt_bf16_kernel<<<(unsigned)CDIV(n / 4, (size_t)256), 256, 0, s>>>(X, Y, n / 4);
  return Y;
}

// transpose convert (activations): (C, I, TT) f32 -> (C, TT, I) bf16
static __bf16* cvt_to_bf16_t(const float* X, int C, int I, int sqrelu, Bump& ws, hipStream_t s) {
  __bf16* Y = (__bf16*)balloc_raw(ws, (size_t)C * I * TT * 2);
  dim3 g(TT / 32, I / 32, C);
  cvt_bf16_t_kernel<<<g, 256, 0, s>>>(X, Y, I, sqrelu);
  return Y;
}

// converts W to bf16 (scratch, freed after enqueue; safe: stream-ordered reuse)
static void run_gemm(const float* W, const __bf16* Xtb, const float* R, float* Out,
                     int C, int O, int I, int flags, Bump& ws, hipStream_t s) {
  size_t mark = ws.off;
  __bf16* Wb = cvt_to_bf16(W, (size_t)C * O * I, ws, s);
  dim3 g(TT / 64, O / 32, C);    // O, I always multiples of 32 in this model
  gemm_mcl_wmma<<<g, 256, 0, s>>>(Wb, Xtb, R, Out, O, I, flags);
  ws.off = mark;
}

static void fte_forward(const float* xin, float* out, const FteP& p, int C, int F,
                        Bump& ws, hipStream_t s) {
  const size_t n = (size_t)C * F * TT;
  const int hd = F / HEADS;
  size_t mark = ws.off;
  float* z  = balloc(ws, n);
  float* q  = balloc(ws, n);
  float* k  = balloc(ws, n);
  float* v  = balloc(ws, n);
  float* a  = balloc(ws, n);
  float* x1 = balloc(ws, n);
  float* hb = balloc(ws, 4 * n);
  float* sc = balloc(ws, (size_t)HEADS * TT * TT);

  copy_f32<<<(unsigned)CDIV(n, (size_t)256), 256, 0, s>>>(xin, out, n);       // idt half of concat
  frame_norm_kernel<<<(unsigned)CDIV(C * TT, 256), 256, 0, s>>>(xin, p.n1w, p.n1b, z, C, F);
  __bf16* zb = cvt_to_bf16_t(z, C, F, 0, ws, s);                              // shared by q,k,v
  run_gemm(p.qw, zb, nullptr, q, C, F, F, 0, ws, s);
  run_gemm(p.kw, zb, nullptr, k, C, F, F, 0, ws, s);
  run_gemm(p.vw, zb, nullptr, v, C, F, F, 0, ws, s);
  size_t pr = (size_t)C * (F / 2) * TT;
  rotary_kernel<<<(unsigned)CDIV(pr, (size_t)256), 256, 0, s>>>(q, C, F, hd);
  rotary_kernel<<<(unsigned)CDIV(pr, (size_t)256), 256, 0, s>>>(k, C, F, hd);
  float scale = 1.0f / sqrtf((float)F);
  for (int c = 0; c < C; ++c) {
    size_t off = (size_t)c * F * TT;
    attn_qk_wmma<<<dim3(TT / 16, TT / 16, HEADS), 32, 0, s>>>(q + off, k + off, sc, hd, scale);
    softmax_rows<<<HEADS * TT, 256, 0, s>>>(sc);
    attn_pv_wmma<<<dim3(TT / 16, (unsigned)CDIV(hd, 16), HEADS), 32, 0, s>>>(sc, v + off, a + off, hd);
  }
  __bf16* ab = cvt_to_bf16_t(a, C, F, 0, ws, s);
  run_gemm(p.ow, ab, xin, x1, C, F, F, FLAG_RESID, ws, s);                    // x1 = x + attn
  frame_norm_kernel<<<(unsigned)CDIV(C * TT, 256), 256, 0, s>>>(x1, p.n2w, p.n2b, z, C, F);
  __bf16* z2b = cvt_to_bf16_t(z, C, F, 0, ws, s);
  run_gemm(p.c1w, z2b, nullptr, hb, C, 4 * F, F, 0, ws, s);
  __bf16* hbb = cvt_to_bf16_t(hb, C, 4 * F, 1, ws, s);                        // fused sqrelu
  run_gemm(p.c2w, hbb, x1, out + n, C, F, 4 * F, FLAG_RESID, ws, s);
  ws.off = mark;
}

static void resblock_forward(const float* xin, float* out, const ResP& p,
                             int Cin, int Cout, int H, bool ds, Bump& ws, hipStream_t s) {
  int sH = ds ? 2 : 1;
  int Ho = (H - 1) / sH + 1;
  size_t mark = ws.off;
  float* zz = balloc(ws, (size_t)Cin * H * TT);
  float* h1 = balloc(ws, (size_t)Cout * H * TT);
  float* h2 = balloc(ws, (size_t)Cout * Ho * TT);
  frame_norm_kernel<<<(unsigned)CDIV(Cin * TT, 256), 256, 0, s>>>(xin, p.nw, p.nb, zz, Cin, H);
  conv3x3_kernel<<<(unsigned)CDIV(Cout * H * TT, 256), 256, 0, s>>>(zz, p.c1, h1, Cin, Cout, H, H, 1, 0);
  conv3x3_kernel<<<(unsigned)CDIV(Cout * Ho * TT, 256), 256, 0, s>>>(h1, p.c2, h2, Cout, Cout, H, Ho, sH, 1);
  res_combine_kernel<<<(unsigned)CDIV(Cout * Ho * TT, 256), 256, 0, s>>>(xin, p.idw, h2, out, Cin, Cout, H, Ho, sH);
  ws.off = mark;
}

extern "C" void kernel_launch(void* const* d_in, const int* in_sizes, int n_in,
                              void* d_out, int out_size, void* d_ws, size_t ws_size,
                              hipStream_t stream) {
  (void)in_sizes; (void)n_in; (void)out_size; (void)ws_size;
  int idx = 0;
  const float* x = (const float*)d_in[idx++];
  auto nextp = [&]() { return (const float*)d_in[idx++]; };
  auto read_fte = [&](FteP& p) { p.n1w=nextp(); p.n1b=nextp(); p.qw=nextp(); p.kw=nextp();
                                 p.vw=nextp(); p.ow=nextp(); p.n2w=nextp(); p.n2b=nextp();
                                 p.c1w=nextp(); p.c2w=nextp(); };
  auto read_res = [&](ResP& p) { p.nw=nextp(); p.nb=nextp(); p.c1=nextp(); p.c2=nextp(); p.idw=nextp(); };

  FteP t0,t1,t2,t3,t4,t5,t6,dt5,dt4,dt3,dt2,dt1;
  ResP e1,e2,e3,e4,e5,e6,d5,d4,d3,d2,d1;
  read_fte(t0); read_res(e1); read_fte(t1); read_res(e2); read_fte(t2);
  read_res(e3); read_fte(t3); read_res(e4); read_fte(t4); read_res(e5); read_fte(t5);
  read_res(e6); read_fte(t6);
  read_res(d5); read_fte(dt5); read_res(d4); read_fte(dt4); read_res(d3); read_fte(dt3);
  read_res(d2); read_fte(dt2); read_res(d1); read_fte(dt1);
  const float* pout = nextp();

  Bump ws{ (char*)d_ws, 0 };

  // ---------------- encoder ----------------
  float* E0 = balloc(ws, (size_t)4 * 1024 * TT);
  fte_forward(x, E0, t0, 2, 1024, ws, stream);
  float* R1 = balloc(ws, (size_t)2 * 1024 * TT);
  resblock_forward(E0, R1, e1, 4, 2, 1024, false, ws, stream);
  float* E1 = balloc(ws, (size_t)4 * 1024 * TT);
  fte_forward(R1, E1, t1, 2, 1024, ws, stream);
  float* R2 = balloc(ws, (size_t)4 * 512 * TT);
  resblock_forward(E1, R2, e2, 4, 4, 1024, true, ws, stream);
  float* E2 = balloc(ws, (size_t)8 * 512 * TT);
  fte_forward(R2, E2, t2, 4, 512, ws, stream);
  float* R3 = balloc(ws, (size_t)8 * 256 * TT);
  resblock_forward(E2, R3, e3, 8, 8, 512, true, ws, stream);
  float* E3 = balloc(ws, (size_t)16 * 256 * TT);
  fte_forward(R3, E3, t3, 8, 256, ws, stream);
  float* R4 = balloc(ws, (size_t)12 * 128 * TT);
  resblock_forward(E3, R4, e4, 16, 12, 256, true, ws, stream);
  float* E4 = balloc(ws, (size_t)24 * 128 * TT);
  fte_forward(R4, E4, t4, 12, 128, ws, stream);
  float* R5 = balloc(ws, (size_t)16 * 64 * TT);
  resblock_forward(E4, R5, e5, 24, 16, 128, true, ws, stream);
  float* E5 = balloc(ws, (size_t)32 * 64 * TT);
  fte_forward(R5, E5, t5, 16, 64, ws, stream);
  float* R6 = balloc(ws, (size_t)20 * 32 * TT);
  resblock_forward(E5, R6, e6, 32, 20, 64, true, ws, stream);
  float* E6 = balloc(ws, (size_t)40 * 32 * TT);
  fte_forward(R6, E6, t6, 20, 32, ws, stream);

  // ---------------- decoder ----------------
  auto dec = [&](const float* hin, int Ch, int Hh, const float* skip, int Cs,
                 const ResP& rp, const FteP& tp, int Cr) -> float* {
    int Hu = 2 * Hh;                 // == F of this decoder stage
    float* out = balloc(ws, (size_t)2 * Cr * Hu * TT);
    size_t mark = ws.off;
    int Ccat = Ch + Cs;
    float* cat = balloc(ws, (size_t)Ccat * Hu * TT);
    upsample_h_kernel<<<(unsigned)CDIV(Ch * Hu * TT, 256), 256, 0, stream>>>(hin, cat, Ch, Hh);
    copy_f32<<<(unsigned)CDIV((size_t)Cs * Hu * TT, (size_t)256), 256, 0, stream>>>(
        skip, cat + (size_t)Ch * Hu * TT, (size_t)Cs * Hu * TT);
    float* rout = balloc(ws, (size_t)Cr * Hu * TT);
    resblock_forward(cat, rout, rp, Ccat, Cr, Hu, false, ws, stream);
    fte_forward(rout, out, tp, Cr, Hu, ws, stream);
    ws.off = mark;
    return out;
  };

  float* H5 = dec(E6, 40, 32,  E5, 32, d5, dt5, 16);
  float* H4 = dec(H5, 32, 64,  E4, 24, d4, dt4, 12);
  float* H3 = dec(H4, 24, 128, E3, 16, d3, dt3, 8);
  float* H2 = dec(H3, 16, 256, E2, 8,  d2, dt2, 4);
  float* H1 = dec(H2, 8,  512, E1, 4,  d1, dt1, 2);

  // final projection (2,4) x (4,1024,512) + edge pad to (2,1025,512)
  out_proj_pad_kernel<<<(unsigned)CDIV(2 * 1025 * TT, 256), 256, 0, stream>>>(
      H1, pout, (float*)d_out, 4, 1024);
}
